// SpatialSelfAttention3D_85968065397289
// MI455X (gfx1250) — matrix operand order repaired and checked by reading, not compiled, not run on hardware
//
#include <hip/hip_runtime.h>

typedef __attribute__((ext_vector_type(16))) _Float16 v16h;
typedef __attribute__((ext_vector_type(8)))  _Float16 v8h;
typedef __attribute__((ext_vector_type(8)))  float    v8f;
typedef __attribute__((ext_vector_type(4)))  unsigned int u32x4;
typedef __attribute__((ext_vector_type(8)))  int          i32x8;
typedef __attribute__((ext_vector_type(4)))  int          i32x4;

#define QTOK 22887          // total tokens = sum of level volumes
#define DM   256            // model dim
#define NC   1280           // 256 (v) + 768 (off) + 256 (attn logits)
#define MTILES ((QTOK + 15) / 16)   // 1431
#define MB     ((MTILES + 7) / 8)   // 179 m-blocks (8 m-tiles per block)
#define LDS_STRIDE 264              // 256 K halves + 8 pad halves -> bank-conflict-free ds_load_b128

// workspace layout (all 64B aligned)
static constexpr size_t OFF_QH    = 0;                                        // Q*256 f16
static constexpr size_t OFF_W1T   = OFF_QH    + (size_t)QTOK * DM * 2;        // 1280*256 f16 (N-major)
static constexpr size_t OFF_WOUTT = OFF_W1T   + (size_t)NC * DM * 2;          // 256*256 f16 (N-major)
static constexpr size_t OFF_BIAS  = OFF_WOUTT + (size_t)DM * DM * 2;          // 1280 f32
static constexpr size_t OFF_INTER = OFF_BIAS  + (size_t)NC * 4 + 192;         // Q*1280 f16
static constexpr size_t OFF_SAM   = OFF_INTER + (size_t)QTOK * NC * 2;        // Q*256 f16

// ---------------- f32 -> f16 convert of query ----------------
__global__ __launch_bounds__(256) void cvt_kernel(const float* __restrict__ q,
                                                  _Float16* __restrict__ qh) {
  size_t i = (size_t)blockIdx.x * 256 + threadIdx.x;
  if (i < (size_t)QTOK * DM) qh[i] = (_Float16)q[i];
}

// ---------------- fold + transpose weights to f16, concat bias ----------------
__global__ __launch_bounds__(256) void prep_kernel(const float* __restrict__ Wv,
                                                   const float* __restrict__ Wo,
                                                   const float* __restrict__ Wa,
                                                   const float* __restrict__ Wout,
                                                   const float* __restrict__ bv,
                                                   const float* __restrict__ bo,
                                                   const float* __restrict__ ba,
                                                   _Float16* __restrict__ W1T,
                                                   _Float16* __restrict__ WoutT,
                                                   float* __restrict__ biascat) {
  int tid = blockIdx.x * 256 + threadIdx.x;
  if (tid < NC * DM) {
    int n = tid / DM, k = tid % DM;
    float val;
    if (n < 256)            val = Wv[k * 256 + n];
    else if (n < 1024) { int j = n - 256;  val = Wo[k * 768 + j] + Wo[(k + 256) * 768 + j]; }
    else               { int j = n - 1024; val = Wa[k * 256 + j] + Wa[(k + 256) * 256 + j]; }
    W1T[(size_t)n * DM + k] = (_Float16)val;
  }
  if (tid < DM * DM) {
    int n = tid / DM, k = tid % DM;
    WoutT[(size_t)n * DM + k] = (_Float16)Wout[k * 256 + n];
  }
  if (tid < NC)
    biascat[tid] = (tid < 256) ? bv[tid] : (tid < 1024 ? bo[tid - 256] : ba[tid - 1024]);
}

// ---------------- stage one 64-col x 256-K f16 weight panel (32KB) into LDS ----------------
// LDS layout: column n at ldsB + n*LDS_STRIDE (8-half pad per column).
// TDM path: one descriptor, 1 row of 4096 8-byte units, pad 4 DWORDs every 128 DWORDs
// so the DMA writes the padded layout directly. Fallback: coalesced manual copy.
__device__ __forceinline__ void stage_panel(_Float16* ldsB, const _Float16* gsrc) {
#if __has_builtin(__builtin_amdgcn_tensor_load_to_lds) && __has_builtin(__builtin_amdgcn_s_wait_tensorcnt)
  if (threadIdx.x < 32) {          // one wave issues the DMA (TDM ignores EXEC)
    const unsigned          lds_addr = (unsigned)(uintptr_t)ldsB;          // low 32 bits = LDS byte offset
    const unsigned long long ga      = (unsigned long long)(uintptr_t)gsrc;
    u32x4 g0; i32x8 g1; i32x4 g2, g3; i32x8 g4;
    // D# group0: count=1 (valid), lds_addr, global_addr[56:0], type=2 ("image")
    g0[0] = 1u;
    g0[1] = lds_addr;
    g0[2] = (unsigned)(ga & 0xffffffffu);
    g0[3] = (unsigned)((ga >> 32) & 0x01ffffffu) | (2u << 30);
    // D# group1: wg_mask=0, data_size=3 (8B units), pad_enable, pad_interval=6 (128 DW = 512B),
    //            pad_amount=3 (4 DW = 16B); tensor_dim0=4096, tensor_dim1=1; tile_dim0=4096;
    //            tensor_dim0_stride=4096
    g1[0] = (3 << 16) | (1 << 20) | (6 << 22) | (3 << 25);
    g1[1] = (int)(4096u << 16);      // tensor_dim0[15:0] @ bits 63:48
    g1[2] = (int)(1u << 16);         // tensor_dim0[31:16]=0 | tensor_dim1[15:0]=1
    g1[3] = (int)(4096u << 16);      // tensor_dim1[31:16]=0 | tile_dim0=4096
    g1[4] = 0;                       // tile_dim1=0 (unused), tile_dim2=0
    g1[5] = 4096;                    // tensor_dim0_stride[31:0]
    g1[6] = 0;                       // stride hi / dim1_stride lo
    g1[7] = 0;
    // D# group2/3: higher dims degenerate (=1), strides 0
    g2[0] = 1; g2[1] = 1; g2[2] = 0; g2[3] = 0;          // tensor_dim2=1, tensor_dim3=1
    g3[0] = 0; g3[1] = (int)(1u << 16); g3[2] = 0; g3[3] = 0;  // tensor_dim4=1
    // this toolchain's builtin is the 6-arg form: extra int32x8 operand
    // (uncharacterized on this lane; zero-filled is the probe-verified form)
    g4[0] = 0; g4[1] = 0; g4[2] = 0; g4[3] = 0; g4[4] = 0; g4[5] = 0; g4[6] = 0; g4[7] = 0;
    __builtin_amdgcn_tensor_load_to_lds(g0, g1, g2, g3, g4, 0);
    __builtin_amdgcn_s_wait_tensorcnt(0);
  }
#else
  for (int c = threadIdx.x; c < (64 * 256) / 8; c += 256) {
    const int n = c >> 5, j = (c & 31) * 8;
    *(v8h*)(ldsB + n * LDS_STRIDE + j) = *(const v8h*)(gsrc + n * 256 + j);
  }
#endif
}

// ---------------- WMMA fragment helpers (gfx1250 16x16x32 f16 layouts) ----------------
// A (16x32, M x K): lane L: m=L&15, g=L>>4. halves h -> K=16g+h
__device__ __forceinline__ v16h load_a_frag(const _Float16* arow, int k0, int g) {
  v8h lo = *(const v8h*)(arow + k0 + 16 * g);
  v8h hi = *(const v8h*)(arow + k0 + 16 * g + 8);
  return __builtin_shufflevector(lo, hi, 0, 1, 2, 3, 4, 5, 6, 7, 8, 9, 10, 11, 12, 13, 14, 15);
}
// B (32x16, K x N): lane L: n=L&15, g=L>>4, halves h -> K = 16g + h (contiguous in LDS column)
__device__ __forceinline__ v16h load_b_frag(const _Float16* bcol) {
  v8h lo = *(const v8h*)(bcol);
  v8h hi = *(const v8h*)(bcol + 8);
  return __builtin_shufflevector(lo, hi, 0, 1, 2, 3, 4, 5, 6, 7, 8, 9, 10, 11, 12, 13, 14, 15);
}

// ---------------- fused input GEMM: inter = query @ [Wv | Wo_fold | Wa_fold] + bias ----------------
// block = 8 waves: 8 m-tiles x one 64-col n-group; B panel staged once in LDS via TDM
__global__ __launch_bounds__(256) void gemm_qw_kernel(const _Float16* __restrict__ Qh,
                                                      const _Float16* __restrict__ W1T,
                                                      const float* __restrict__ biascat,
                                                      _Float16* __restrict__ inter) {
  __shared__ __align__(128) _Float16 ldsB[64 * LDS_STRIDE];
  const int NG = NC / 64;                    // 20 n-groups
  const int mblock = blockIdx.x / NG;
  const int n0     = (blockIdx.x % NG) * 64;
  stage_panel(ldsB, W1T + (size_t)n0 * DM);
  __syncthreads();

  const int lane  = threadIdx.x & 31;
  const int mtile = mblock * 8 + (threadIdx.x >> 5);
  if (mtile >= MTILES) return;
  const int g  = lane >> 4;
  const int lm = lane & 15;
  int arowi = mtile * 16 + lm; if (arowi >= QTOK) arowi = QTOK - 1;   // clamp pad rows
  const _Float16* arow = Qh + (size_t)arowi * DM;

  v8f c[4] = {v8f{}, v8f{}, v8f{}, v8f{}};
  #pragma unroll
  for (int ks = 0; ks < 8; ++ks) {
    const int k0 = ks * 32;
    v16h a = load_a_frag(arow, k0, g);
    #pragma unroll
    for (int t = 0; t < 4; ++t) {
      v16h b = load_b_frag(ldsB + (t * 16 + lm) * LDS_STRIDE + k0 + 16 * g);
      c[t] = __builtin_amdgcn_wmma_f32_16x16x32_f16(false, a, false, b, (short)0, c[t], false, false);
    }
  }
  #pragma unroll
  for (int t = 0; t < 4; ++t) {
    const int col = n0 + t * 16 + lm;
    const float bias = biascat[col];
    #pragma unroll
    for (int r = 0; r < 8; ++r) {                 // C/D: VGPR r -> M = r + 8g, N = lane&15
      const int orow = mtile * 16 + 8 * g + r;
      if (orow < QTOK) inter[(size_t)orow * NC + col] = (_Float16)(c[t][r] + bias);
    }
  }
}

// ---------------- sampling: one wave per (q, head); lane = channel (hd=32=wave32) ----------------
__global__ __launch_bounds__(256) void sample_kernel(const _Float16* __restrict__ inter,
                                                     const float* __restrict__ refpts,
                                                     _Float16* __restrict__ sam) {
  const int lane = threadIdx.x & 31;
  const int wave = blockIdx.x * 8 + (threadIdx.x >> 5);
  if (wave >= QTOK * 8) return;
  const int q = wave >> 3;
  const int h = wave & 7;

  const int Hs[4] = {50, 25, 13, 7};
  const int Ws[4] = {50, 25, 13, 7};
  const int Ds[4] = {8, 4, 2, 1};
  const int St[4] = {0, 20000, 22500, 22838};

  const _Float16* qrow = inter + (size_t)q * NC;
  float acc = 0.f;
  #pragma unroll
  for (int nq = 0; nq < 2; ++nq) {
    // softmax over 16 (L*P) logits for this (q, h, nq)
    const _Float16* lptr = qrow + 1024 + (h * 2 + nq) * 16;
    float lg[16];
    float mx = -1e30f;
    #pragma unroll
    for (int i = 0; i < 16; ++i) { lg[i] = (float)lptr[i]; mx = fmaxf(mx, lg[i]); }
    float s = 0.f;
    #pragma unroll
    for (int i = 0; i < 16; ++i) { lg[i] = __expf(lg[i] - mx); s += lg[i]; }
    const float inv = 0.5f / s;                 // fold mean over NQ=2
    const float* rp = refpts + ((size_t)nq * QTOK + q) * 12;
    #pragma unroll
    for (int l = 0; l < 4; ++l) {
      const int W = Ws[l], H = Hs[l], D = Ds[l], start = St[l];
      const float rx = rp[l * 3 + 0] * (float)W;
      const float ry = rp[l * 3 + 1] * (float)H;
      const float rz = rp[l * 3 + 2] * (float)D;
      #pragma unroll
      for (int p = 0; p < 4; ++p) {
        const _Float16* optr = qrow + 256 + ((((h * 2 + nq) * 4 + l) * 4 + p) * 3);
        // ref*W + (off/W)*W - 0.5 = ref*W + off - 0.5 : offset already in grid units
        const float x = rx + (float)optr[0] - 0.5f;
        const float y = ry + (float)optr[1] - 0.5f;
        const float z = rz + (float)optr[2] - 0.5f;
        const float xf = floorf(x), yf = floorf(y), zf = floorf(z);
        const float fx = x - xf, fy = y - yf, fz = z - zf;
        const int x0 = (int)xf, y0 = (int)yf, z0 = (int)zf;
        // early prefetch of the base corner cacheline (global_prefetch_b8)
        if ((x0 >= 0) & (x0 < W) & (y0 >= 0) & (y0 < H) & (z0 >= 0) & (z0 < D)) {
          const int tok0 = start + (y0 * W + x0) * D + z0;
          __builtin_prefetch(inter + (size_t)tok0 * NC + h * 32 + lane, 0, 1);
        }
        const float aw = lg[l * 4 + p] * inv;
        #pragma unroll
        for (int dx = 0; dx < 2; ++dx)
          #pragma unroll
          for (int dy = 0; dy < 2; ++dy)
            #pragma unroll
            for (int dz = 0; dz < 2; ++dz) {
              const int xi = x0 + dx, yi = y0 + dy, zi = z0 + dz;
              if ((xi >= 0) & (xi < W) & (yi >= 0) & (yi < H) & (zi >= 0) & (zi < D)) {
                const float w = (dx ? fx : 1.f - fx) * (dy ? fy : 1.f - fy) * (dz ? fz : 1.f - fz);
                const int tok = start + (yi * W + xi) * D + zi;
                const float v = (float)inter[(size_t)tok * NC + h * 32 + lane];
                acc = fmaf(aw * w, v, acc);
              }
            }
      }
    }
  }
  sam[(size_t)q * DM + h * 32 + lane] = (_Float16)acc;
}

// ---------------- output GEMM: out = sam @ Wout + bout + query (residual) ----------------
__global__ __launch_bounds__(256) void gemm_out_kernel(const _Float16* __restrict__ sam,
                                                       const _Float16* __restrict__ WoutT,
                                                       const float* __restrict__ bout,
                                                       const float* __restrict__ query,
                                                       float* __restrict__ out) {
  __shared__ __align__(128) _Float16 ldsB[64 * LDS_STRIDE];
  const int mblock = blockIdx.x / 4;                   // 256 cols = 4 n-groups
  const int n0     = (blockIdx.x % 4) * 64;
  stage_panel(ldsB, WoutT + (size_t)n0 * DM);
  __syncthreads();

  const int lane  = threadIdx.x & 31;
  const int mtile = mblock * 8 + (threadIdx.x >> 5);
  if (mtile >= MTILES) return;
  const int g  = lane >> 4;
  const int lm = lane & 15;
  int arowi = mtile * 16 + lm; if (arowi >= QTOK) arowi = QTOK - 1;
  const _Float16* arow = sam + (size_t)arowi * DM;

  v8f c[4] = {v8f{}, v8f{}, v8f{}, v8f{}};
  #pragma unroll
  for (int ks = 0; ks < 8; ++ks) {
    const int k0 = ks * 32;
    v16h a = load_a_frag(arow, k0, g);
    #pragma unroll
    for (int t = 0; t < 4; ++t) {
      v16h b = load_b_frag(ldsB + (t * 16 + lm) * LDS_STRIDE + k0 + 16 * g);
      c[t] = __builtin_amdgcn_wmma_f32_16x16x32_f16(false, a, false, b, (short)0, c[t], false, false);
    }
  }
  #pragma unroll
  for (int t = 0; t < 4; ++t) {
    const int col = n0 + t * 16 + lm;
    const float bias = bout[col];
    #pragma unroll
    for (int r = 0; r < 8; ++r) {
      const int orow = mtile * 16 + 8 * g + r;
      if (orow < QTOK)
        out[(size_t)orow * DM + col] = c[t][r] + bias + query[(size_t)orow * DM + col];
    }
  }
}

extern "C" void kernel_launch(void* const* d_in, const int* in_sizes, int n_in,
                              void* d_out, int out_size, void* d_ws, size_t ws_size,
                              hipStream_t stream) {
  const float* query  = (const float*)d_in[0];
  const float* refpts = (const float*)d_in[1];
  // d_in[2] spatial_shapes, d_in[3] level_start_index: compile-time constants, hardcoded
  const float* Wv   = (const float*)d_in[4];
  const float* bv   = (const float*)d_in[5];
  const float* Wo   = (const float*)d_in[6];
  const float* bo   = (const float*)d_in[7];
  const float* Wa   = (const float*)d_in[8];
  const float* ba   = (const float*)d_in[9];
  const float* Wout = (const float*)d_in[10];
  const float* bout = (const float*)d_in[11];

  char* ws = (char*)d_ws;
  _Float16* Qh      = (_Float16*)(ws + OFF_QH);
  _Float16* W1T     = (_Float16*)(ws + OFF_W1T);
  _Float16* WoutT   = (_Float16*)(ws + OFF_WOUTT);
  float*    biascat = (float*)   (ws + OFF_BIAS);
  _Float16* inter   = (_Float16*)(ws + OFF_INTER);
  _Float16* sam     = (_Float16*)(ws + OFF_SAM);
  float*    out     = (float*)d_out;

  // 1) convert query to f16
  cvt_kernel<<<(unsigned)(((size_t)QTOK * DM + 255) / 256), 256, 0, stream>>>(query, Qh);
  // 2) fold/transpose weights + concat bias
  prep_kernel<<<(NC * DM + 255) / 256, 256, 0, stream>>>(Wv, Wo, Wa, Wout, bv, bo, ba,
                                                         W1T, WoutT, biascat);
  // 3) fused input GEMM (Qx256)@(256x1280) -> inter (v | off | attn); TDM-staged B panels
  gemm_qw_kernel<<<MB * (NC / 64), 256, 0, stream>>>(Qh, W1T, biascat, inter);
  // 4) softmax + trilinear deformable sampling, mean over NQ folded in
  sample_kernel<<<QTOK, 256, 0, stream>>>(inter, refpts, sam);
  // 5) output GEMM + bias + residual; TDM-staged B panels
  gemm_out_kernel<<<MB * 4, 256, 0, stream>>>(sam, WoutT, bout, query, out);
}
